// HeteroKANGuard_91242285236636
// MI455X (gfx1250) — compile-verified, matrix operand order checked by reading, not run on hardware
//
#include <hip/hip_runtime.h>
#include <hip/hip_bf16.h>

typedef __attribute__((ext_vector_type(16))) __bf16 v16bf;
typedef __attribute__((ext_vector_type(8)))  float  v8f;

#define NE_  200000
#define NU_  100000
#define EE_  1000000
#define FIN_ 64
#define H_   128
#define NB_  8

// ---------- WMMA fragment index helpers (CDNA5 16x16x32 bf16 layouts) ----------
// A (16x32, MxK): lane = row + ((k&8)?16:0); elem = (k&7) + ((k&16)?8:0)
__device__ __forceinline__ void a_slot(int row, int k, int& kt, int& slot) {
    kt = k >> 5;
    int kk = k & 31;
    int lane = row + ((kk & 8) ? 16 : 0);
    int e    = (kk & 7) + ((kk & 16) ? 8 : 0);
    slot = lane * 16 + e;
}

// B (32x16, KxN): lanes 0-15 hold K=0..7,16..23 (elem order k=e / 16+(e-8)),
//                 lanes 16-31 hold K=8..15,24..31 (k=8+e / 24+(e-8)); col = lane&15.
__device__ __forceinline__ int b_k_of(int lane, int e) {
    if (lane < 16) return (e < 8) ? e : (16 + (e - 8));
    else           return (e < 8) ? (8 + e) : (24 + (e - 8));
}

// ---------- weight pre-swizzle into per-lane B fragments (bf16) ----------
// mode 0: plain W[N,K] (Y = X @ W^T)
// mode 1: augmented KAN weight: k<Fin -> base[n,k]; else spline[n,i,kk]*scaler[n,i]
__global__ void pack_b_kernel(const float* __restrict__ W,
                              const float* __restrict__ spline,
                              const float* __restrict__ scaler,
                              __bf16* __restrict__ out,
                              int Ktiles, int Ntiles, int K, int Fin, int Nreal, int mode)
{
    int idx = blockIdx.x * blockDim.x + threadIdx.x;
    int total = Ktiles * Ntiles * 32;
    if (idx >= total) return;
    int lane = idx & 31;
    int tile = idx >> 5;
    int nt = tile % Ntiles;
    int kt = tile / Ntiles;
    int n  = nt * 16 + (lane & 15);
    __bf16* dst = out + (size_t)(tile * 32 + lane) * 16;
#pragma unroll
    for (int e = 0; e < 16; ++e) {
        int k = kt * 32 + b_k_of(lane, e);
        float v = 0.0f;
        if (n < Nreal && k < K) {
            if (mode == 0) {
                v = W[(size_t)n * K + k];
            } else {
                if (k < Fin) v = W[(size_t)n * Fin + k];
                else {
                    int i  = (k - Fin) >> 3;
                    int kk = (k - Fin) & 7;
                    v = spline[((size_t)n * Fin + i) * 8 + kk] * scaler[(size_t)n * Fin + i];
                }
            }
        }
        dst[e] = (__bf16)v;
    }
}

// ---------- xe = x_email @ w_email^T + b  (M=NE, K=64, N=128) ----------
__global__ void email_lin_kernel(const float* __restrict__ X, const __bf16* __restrict__ pW,
                                 const float* __restrict__ bias, float* __restrict__ Y)
{
    __shared__ __align__(32) __bf16 A[2][512];
    int tid  = threadIdx.x;
    int row0 = blockIdx.x * 16;
    for (int i = tid; i < 16 * 64; i += 256) {
        int row = i >> 6, k = i & 63;
        int kt, slot; a_slot(row, k, kt, slot);
        A[kt][slot] = (__bf16)X[(size_t)(row0 + row) * 64 + k];
    }
    __syncthreads();
    int wave = tid >> 5, lane = tid & 31, nt = wave;
    v8f acc = {};
#pragma unroll
    for (int kt = 0; kt < 2; ++kt) {
        v16bf a = *(const v16bf*)&A[kt][lane * 16];
        v16bf b = *(const v16bf*)(pW + (size_t)((kt * 8 + nt) * 32 + lane) * 16);
        acc = __builtin_amdgcn_wmma_f32_16x16x32_bf16(false, a, false, b, (short)0, acc, false, false);
    }
    int n = nt * 16 + (lane & 15);
    float bv = bias[n];
#pragma unroll
    for (int r = 0; r < 8; ++r) {
        int m = r + ((lane >= 16) ? 8 : 0);
        Y[(size_t)(row0 + m) * H_ + n] = acc[r] + bv;
    }
}

// ---------- edge kernels (segment mean pieces) ----------
__global__ void count_kernel(const int* __restrict__ dst, float* __restrict__ cnt, int E)
{
    int i = blockIdx.x * blockDim.x + threadIdx.x;
    if (i < E) atomicAdd(&cnt[dst[i]], 1.0f);
}

__global__ void scatter_kernel(const float* __restrict__ Xsrc, const int* __restrict__ src,
                               const int* __restrict__ dst, float* __restrict__ agg, int E)
{
    int i = blockIdx.x * blockDim.x + threadIdx.x;
    if (i >= E * 32) return;               // 32 = H/4 chunks of float4
    int e = i >> 5, c = (i & 31) << 2;
    int s = src[e], d = dst[e];
    const float4 v = *(const float4*)&Xsrc[(size_t)s * H_ + c];
    float* a = &agg[(size_t)d * H_ + c];
    atomicAdd(a + 0, v.x); atomicAdd(a + 1, v.y);
    atomicAdd(a + 2, v.z); atomicAdd(a + 3, v.w);
}

// ---------- Y = relu(mean @ Wn^T + b + xdst @ Wr^T)  (K=128, N=128) ----------
__global__ void sage_finish_kernel(const float* __restrict__ agg, const float* __restrict__ cnt,
                                   const float* __restrict__ xdst,
                                   const __bf16* __restrict__ pWn, const __bf16* __restrict__ pWr,
                                   const float* __restrict__ bias, float* __restrict__ Y)
{
    __shared__ __align__(32) __bf16 Am[4][512];
    __shared__ __align__(32) __bf16 Ax[4][512];
    int tid  = threadIdx.x;
    int row0 = blockIdx.x * 16;
    for (int i = tid; i < 16 * 128; i += 256) {
        int row = i >> 7, k = i & 127;
        size_t g = (size_t)(row0 + row) * H_ + k;
        float rc = 1.0f / fmaxf(cnt[row0 + row], 1.0f);
        int kt, slot; a_slot(row, k, kt, slot);
        Am[kt][slot] = (__bf16)(agg[g] * rc);
        Ax[kt][slot] = (__bf16)xdst[g];
    }
    __syncthreads();
    int wave = tid >> 5, lane = tid & 31, nt = wave;
    v8f acc = {};
#pragma unroll
    for (int kt = 0; kt < 4; ++kt) {
        v16bf a = *(const v16bf*)&Am[kt][lane * 16];
        v16bf b = *(const v16bf*)(pWn + (size_t)((kt * 8 + nt) * 32 + lane) * 16);
        acc = __builtin_amdgcn_wmma_f32_16x16x32_bf16(false, a, false, b, (short)0, acc, false, false);
    }
#pragma unroll
    for (int kt = 0; kt < 4; ++kt) {
        v16bf a = *(const v16bf*)&Ax[kt][lane * 16];
        v16bf b = *(const v16bf*)(pWr + (size_t)((kt * 8 + nt) * 32 + lane) * 16);
        acc = __builtin_amdgcn_wmma_f32_16x16x32_bf16(false, a, false, b, (short)0, acc, false, false);
    }
    int n = nt * 16 + (lane & 15);
    float bv = bias[n];
#pragma unroll
    for (int r = 0; r < 8; ++r) {
        int m = r + ((lane >= 16) ? 8 : 0);
        Y[(size_t)(row0 + m) * H_ + n] = fmaxf(acc[r] + bv, 0.0f);
    }
}

// ---------- KAN: cubic B-spline bases (grid_size=5, k=3, uniform grid) ----------
__device__ __forceinline__ void compute_bases(float x, float* b8)
{
    const float h = 2.0f / 5.0f;
    float g[12];
#pragma unroll
    for (int t = 0; t < 12; ++t) g[t] = (float)(t - 3) * h - 1.0f;
    float b[11];
#pragma unroll
    for (int t = 0; t < 11; ++t) b[t] = (x >= g[t] && x < g[t + 1]) ? 1.0f : 0.0f;
#pragma unroll
    for (int j = 1; j <= 3; ++j) {
        float inv = 1.0f / ((float)j * h);
#pragma unroll
        for (int t = 0; t < 10; ++t) {
            if (t < 11 - j)
                b[t] = (x - g[t]) * inv * b[t] + (g[t + j + 1] - x) * inv * b[t + 1];
        }
    }
#pragma unroll
    for (int t = 0; t < 8; ++t) b8[t] = b[t];
}

// KAN layer as a single GEMM over phi(x) = [silu(x), bases(x)] of length 9*FIN.
template<int FIN, int KT, int NT, int NOUT, int NREAL, int THREADS>
__global__ void kan_kernel(const float* __restrict__ X, const __bf16* __restrict__ pW,
                           float* __restrict__ Y)
{
    __shared__ __align__(32) __bf16 A[KT * 512];
    int tid  = threadIdx.x;
    int row0 = blockIdx.x * 16;
    for (int i = tid; i < 16 * FIN; i += THREADS) {
        int row = i / FIN, f = i % FIN;
        float x = X[(size_t)(row0 + row) * FIN + f];
        float sig = 1.0f / (1.0f + __expf(-x));
        int kt, slot;
        a_slot(row, f, kt, slot);
        A[kt * 512 + slot] = (__bf16)(x * sig);
        float b8[8];
        compute_bases(x, b8);
#pragma unroll
        for (int kk = 0; kk < 8; ++kk) {
            a_slot(row, FIN + f * 8 + kk, kt, slot);
            A[kt * 512 + slot] = (__bf16)b8[kk];
        }
    }
    __syncthreads();
    int wave = tid >> 5, lane = tid & 31, nt = wave;
    v8f acc = {};
    for (int kt = 0; kt < KT; ++kt) {
        v16bf a = *(const v16bf*)&A[kt * 512 + lane * 16];
        v16bf b = *(const v16bf*)(pW + (size_t)((kt * NT + nt) * 32 + lane) * 16);
        acc = __builtin_amdgcn_wmma_f32_16x16x32_bf16(false, a, false, b, (short)0, acc, false, false);
    }
    int n = nt * 16 + (lane & 15);
    if (n < NREAL) {
#pragma unroll
        for (int r = 0; r < 8; ++r) {
            int m = r + ((lane >= 16) ? 8 : 0);
            Y[(size_t)(row0 + m) * NOUT + n] = acc[r];
        }
    }
}

// ------------------------------------------------------------------------------
extern "C" void kernel_launch(void* const* d_in, const int* in_sizes, int n_in,
                              void* d_out, int out_size, void* d_ws, size_t ws_size,
                              hipStream_t stream)
{
    const float* x_email  = (const float*)d_in[0];
    const int*   ei_ue    = (const int*)d_in[1];   // [2,E]: row0=src(user), row1=dst(email)
    const int*   ei_eu    = (const int*)d_in[2];   // [2,E]: row0=src(email), row1=dst(user)
    const float* w_email  = (const float*)d_in[3];
    const float* b_email  = (const float*)d_in[4];
    const float* emb_user = (const float*)d_in[5];
    const float* w_l1ue_n = (const float*)d_in[6];
    const float* b_l1ue   = (const float*)d_in[7];
    const float* w_l1ue_r = (const float*)d_in[8];
    const float* w_l1eu_n = (const float*)d_in[9];
    const float* b_l1eu   = (const float*)d_in[10];
    const float* w_l1eu_r = (const float*)d_in[11];
    const float* w_l2ue_n = (const float*)d_in[12];
    const float* b_l2ue   = (const float*)d_in[13];
    const float* w_l2ue_r = (const float*)d_in[14];
    const float* k1_base  = (const float*)d_in[18];
    const float* k1_spl   = (const float*)d_in[19];
    const float* k1_scl   = (const float*)d_in[20];
    const float* k2_base  = (const float*)d_in[22];
    const float* k2_spl   = (const float*)d_in[23];
    const float* k2_scl   = (const float*)d_in[24];

    char* ws = (char*)d_ws;
    size_t off = 0;
    auto alloc = [&](size_t bytes) { size_t o = off; off = (off + bytes + 255) & ~(size_t)255; return o; };

    const size_t NEH = (size_t)NE_ * H_ * sizeof(float);
    const size_t NUH = (size_t)NU_ * H_ * sizeof(float);

    float* XE   = (float*)(ws + alloc(NEH));            // xe, later reused for e2
    float* AGG  = (float*)(ws + alloc(NEH));            // scatter scratch, later kan hidden
    float* E1   = (float*)(ws + alloc(NEH));
    float* U1   = (float*)(ws + alloc(NUH));
    float* CNTE = (float*)(ws + alloc((size_t)NE_ * 4));
    float* CNTU = (float*)(ws + alloc((size_t)NU_ * 4));

    auto packsz = [](int kt, int nt) { return (size_t)kt * nt * 512 * sizeof(__bf16); };
    __bf16* pWe    = (__bf16*)(ws + alloc(packsz(2, 8)));
    __bf16* pWn1ue = (__bf16*)(ws + alloc(packsz(4, 8)));
    __bf16* pWr1ue = (__bf16*)(ws + alloc(packsz(4, 8)));
    __bf16* pWn1eu = (__bf16*)(ws + alloc(packsz(4, 8)));
    __bf16* pWr1eu = (__bf16*)(ws + alloc(packsz(4, 8)));
    __bf16* pWn2ue = (__bf16*)(ws + alloc(packsz(4, 8)));
    __bf16* pWr2ue = (__bf16*)(ws + alloc(packsz(4, 8)));
    __bf16* pWaug1 = (__bf16*)(ws + alloc(packsz(36, 4)));
    __bf16* pWaug2 = (__bf16*)(ws + alloc(packsz(18, 1)));

    float* E2 = XE;     // reuse after u1 is done
    float* HB = AGG;    // reuse after last scatter phase

    // ---- weight pre-swizzle (tiny, L2-resident) ----
    auto pack = [&](const float* W, const float* s, const float* sc, __bf16* out,
                    int kt, int nt, int K, int Fin, int Nreal, int mode) {
        int total = kt * nt * 32;
        pack_b_kernel<<<(total + 255) / 256, 256, 0, stream>>>(W, s, sc, out, kt, nt, K, Fin, Nreal, mode);
    };
    pack(w_email,  nullptr, nullptr, pWe,    2, 8, 64,   0,  128, 0);
    pack(w_l1ue_n, nullptr, nullptr, pWn1ue, 4, 8, 128,  0,  128, 0);
    pack(w_l1ue_r, nullptr, nullptr, pWr1ue, 4, 8, 128,  0,  128, 0);
    pack(w_l1eu_n, nullptr, nullptr, pWn1eu, 4, 8, 128,  0,  128, 0);
    pack(w_l1eu_r, nullptr, nullptr, pWr1eu, 4, 8, 128,  0,  128, 0);
    pack(w_l2ue_n, nullptr, nullptr, pWn2ue, 4, 8, 128,  0,  128, 0);
    pack(w_l2ue_r, nullptr, nullptr, pWr2ue, 4, 8, 128,  0,  128, 0);
    pack(k1_base,  k1_spl,  k1_scl,  pWaug1, 36, 4, 1152, 128, 64, 1);
    pack(k2_base,  k2_spl,  k2_scl,  pWaug2, 18, 1, 576,  64,  2,  1);

    // ---- xe = email_lin(x_email) ----
    email_lin_kernel<<<NE_ / 16, 256, 0, stream>>>(x_email, pWe, b_email, XE);

    // ---- degree counts (ei fixed: compute once per launch) ----
    hipMemsetAsync(CNTE, 0, (size_t)NE_ * 4, stream);
    hipMemsetAsync(CNTU, 0, (size_t)NU_ * 4, stream);
    count_kernel<<<(EE_ + 255) / 256, 256, 0, stream>>>(ei_ue + EE_, CNTE, EE_);
    count_kernel<<<(EE_ + 255) / 256, 256, 0, stream>>>(ei_eu + EE_, CNTU, EE_);

    const int sblocks = (EE_ * 32 + 255) / 256;

    // ---- layer 1, email-dst: e1 = relu(sage(emb_user -> xe)) ----
    hipMemsetAsync(AGG, 0, NEH, stream);
    scatter_kernel<<<sblocks, 256, 0, stream>>>(emb_user, ei_ue, ei_ue + EE_, AGG, EE_);
    sage_finish_kernel<<<NE_ / 16, 256, 0, stream>>>(AGG, CNTE, XE, pWn1ue, pWr1ue, b_l1ue, E1);

    // ---- layer 1, user-dst: u1 = relu(sage(xe -> emb_user)) ----
    hipMemsetAsync(AGG, 0, NUH, stream);
    scatter_kernel<<<sblocks, 256, 0, stream>>>(XE, ei_eu, ei_eu + EE_, AGG, EE_);
    sage_finish_kernel<<<NU_ / 16, 256, 0, stream>>>(AGG, CNTU, emb_user, pWn1eu, pWr1eu, b_l1eu, U1);

    // ---- layer 2, email-dst: e2 = relu(sage(u1 -> e1)) ----
    hipMemsetAsync(AGG, 0, NEH, stream);
    scatter_kernel<<<sblocks, 256, 0, stream>>>(U1, ei_ue, ei_ue + EE_, AGG, EE_);
    sage_finish_kernel<<<NE_ / 16, 256, 0, stream>>>(AGG, CNTE, E1, pWn2ue, pWr2ue, b_l2ue, E2);

    // ---- KAN head: h = kan1(e2) [N=64, K=9*128], out = kan2(h) [N=2, K=9*64] ----
    kan_kernel<128, 36, 4, 64, 64, 128><<<NE_ / 16, 128, 0, stream>>>(E2, pWaug1, HB);
    kan_kernel<64, 18, 1, 2, 2, 32><<<NE_ / 16, 32, 0, stream>>>(HB, pWaug2, (float*)d_out);
}
